// Attention_21706764714809
// MI455X (gfx1250) — matrix operand order repaired and checked by reading, not compiled, not run on hardware
//
#include <hip/hip_runtime.h>
#include <math.h>

// Problem constants (match reference).
#define BB 16
#define TT 1024
#define SS 1024
#define EE 1024
#define DD 1024
#define CC (EE + DD)   // concat dim = 2048

typedef __bf16  v16bf __attribute__((ext_vector_type(16)));
typedef float   v8f   __attribute__((ext_vector_type(8)));

union FragBf {
    v16bf v;
    uint4 q[2];
    unsigned int  ui[8];
    unsigned short u[16];
};

__device__ __forceinline__ unsigned short f2bf(float f) {
    unsigned int u = __float_as_uint(f);
    unsigned int r = u + 0x7FFFu + ((u >> 16) & 1u);   // round-to-nearest-even
    return (unsigned short)(r >> 16);
}

// ---------------------------------------------------------------------------
// fp32 -> bf16 bulk convert
// ---------------------------------------------------------------------------
__global__ void k_cvt_bf16(const float* __restrict__ in, unsigned short* __restrict__ out, size_t n) {
    size_t i = (size_t)blockIdx.x * blockDim.x + threadIdx.x;
    size_t stride = (size_t)gridDim.x * blockDim.x;
    for (; i < n; i += stride) out[i] = f2bf(in[i]);
}

// ---------------------------------------------------------------------------
// NT WMMA core: acc += A[M,K](row-major, ld=ldkA) * B[N,K](row-major, ld=ldkB)^T
// Wave computes 2x4 tiles of 16x16 at (mBase, nBase). K multiple of 32,
// compile-time constant at every call site (function is force-inlined) so the
// K loop unrolls by 8 with immediate load offsets off hoisted row pointers.
// Fragment layouts used here (bf16, 16x16x32; contiguous k, same for A and B):
//   A lane(l): row m=l&15, halves 0..15 -> k=kk+16*hi..
//   B lane(l): row n=l&15, halves 0..15 -> k=kk+16*hi..
// ---------------------------------------------------------------------------
__device__ __forceinline__ void gemm_nt_core(const unsigned short* __restrict__ A, int ldkA,
                                             const unsigned short* __restrict__ Bm, int ldkB,
                                             int K, int mBase, int nBase, int lane,
                                             v8f acc[2][4]) {
    const int hi  = (lane >> 4) & 1;
    const int l15 = lane & 15;

    const unsigned short* ap[2];
    const unsigned short* bp[4];
#pragma unroll
    for (int mi = 0; mi < 2; ++mi)
        ap[mi] = A + (size_t)(mBase + mi * 16 + l15) * ldkA + hi * 16;
#pragma unroll
    for (int ni = 0; ni < 4; ++ni)
        bp[ni] = Bm + (size_t)(nBase + ni * 16 + l15) * ldkB + hi * 16;

#pragma unroll 8
    for (int kk = 0; kk < K; kk += 32) {
        // Load ALL fragments of this k-step first (12 clauseable b128 loads),
        // then issue all 8 WMMAs back-to-back.
        FragBf a[2], bfr[4];
#pragma unroll
        for (int mi = 0; mi < 2; ++mi) {
            a[mi].q[0] = *(const uint4*)(ap[mi] + kk);
            a[mi].q[1] = *(const uint4*)(ap[mi] + kk + 8);
        }
#pragma unroll
        for (int ni = 0; ni < 4; ++ni) {
            bfr[ni].q[0] = ((const uint4*)(bp[ni] + kk))[0];
            bfr[ni].q[1] = ((const uint4*)(bp[ni] + kk))[1];
        }
#pragma unroll
        for (int ni = 0; ni < 4; ++ni)
#pragma unroll
            for (int mi = 0; mi < 2; ++mi)
                acc[mi][ni] = __builtin_amdgcn_wmma_f32_16x16x32_bf16(
                    false, a[mi].v, false, bfr[ni].v, (short)0, acc[mi][ni], false, false);
    }
}

// ---------------------------------------------------------------------------
// GEMM1: proj[b,s,d] = sum_e enc[b,s,e]*W_attn[d,e] + b_attn[d]   (store bf16)
// grid: (D/128, S/128, B), block 256
// ---------------------------------------------------------------------------
__global__ void __launch_bounds__(256)
k_gemm_proj(const unsigned short* __restrict__ enc_bf,
            const unsigned short* __restrict__ Wa_bf,
            const float* __restrict__ bias,
            unsigned short* __restrict__ proj_bf) {
    const int b    = blockIdx.z;
    const int tid  = threadIdx.x;
    const int lane = tid & 31, w = tid >> 5;
    const int mBase = blockIdx.y * 128 + (w >> 1) * 32;
    const int nBase = blockIdx.x * 128 + (w & 1) * 64;
    const int hi = (lane >> 4) & 1, l15 = lane & 15;

    v8f zero = {0.f,0.f,0.f,0.f,0.f,0.f,0.f,0.f};
    v8f acc[2][4];
#pragma unroll
    for (int mi = 0; mi < 2; ++mi)
#pragma unroll
        for (int ni = 0; ni < 4; ++ni) acc[mi][ni] = zero;

    gemm_nt_core(enc_bf + (size_t)b * SS * EE, EE, Wa_bf, EE, EE, mBase, nBase, lane, acc);

    unsigned short* out = proj_bf + (size_t)b * SS * DD;
#pragma unroll
    for (int mi = 0; mi < 2; ++mi)
#pragma unroll
        for (int ni = 0; ni < 4; ++ni) {
            int n = nBase + ni * 16 + l15;
            float bv = bias[n];
#pragma unroll
            for (int r = 0; r < 8; ++r) {
                int m = mBase + mi * 16 + hi * 8 + r;
                out[(size_t)m * DD + n] = f2bf(acc[mi][ni][r] + bv);
            }
        }
}

// ---------------------------------------------------------------------------
// GEMM2: energies[b,t,s] = sum_d hidden[b,t,d]*proj[b,s,d]   (store fp32)
// grid: (S/128, T/128, B)
// ---------------------------------------------------------------------------
__global__ void __launch_bounds__(256)
k_gemm_energy(const unsigned short* __restrict__ hid_bf,
              const unsigned short* __restrict__ proj_bf,
              float* __restrict__ energ) {
    const int b    = blockIdx.z;
    const int tid  = threadIdx.x;
    const int lane = tid & 31, w = tid >> 5;
    const int mBase = blockIdx.y * 128 + (w >> 1) * 32;
    const int nBase = blockIdx.x * 128 + (w & 1) * 64;
    const int hi = (lane >> 4) & 1, l15 = lane & 15;

    v8f zero = {0.f,0.f,0.f,0.f,0.f,0.f,0.f,0.f};
    v8f acc[2][4];
#pragma unroll
    for (int mi = 0; mi < 2; ++mi)
#pragma unroll
        for (int ni = 0; ni < 4; ++ni) acc[mi][ni] = zero;

    gemm_nt_core(hid_bf + (size_t)b * TT * DD, DD,
                 proj_bf + (size_t)b * SS * DD, DD, DD, mBase, nBase, lane, acc);

    float* out = energ + (size_t)b * TT * SS;
#pragma unroll
    for (int mi = 0; mi < 2; ++mi)
#pragma unroll
        for (int ni = 0; ni < 4; ++ni) {
            int n = nBase + ni * 16 + l15;
#pragma unroll
            for (int r = 0; r < 8; ++r) {
                int m = mBase + mi * 16 + hi * 8 + r;
                out[(size_t)m * SS + n] = acc[mi][ni][r];
            }
        }
}

// ---------------------------------------------------------------------------
// Softmax over S per (b,t) row. One 256-thread block per row.
// Writes fp32 weights (output #2) and bf16 copy (for GEMM3 A operand).
// ---------------------------------------------------------------------------
__global__ void __launch_bounds__(256)
k_softmax(const float* __restrict__ energ,
          float* __restrict__ wts,
          unsigned short* __restrict__ wbf) {
    const int row = blockIdx.x;              // b*T + t
    const int tid = threadIdx.x;
    const float* e = energ + (size_t)row * SS;
    __shared__ float red[256];

    float v[4];
    float m = -1e30f;
#pragma unroll
    for (int i = 0; i < 4; ++i) { v[i] = e[tid + i * 256]; m = fmaxf(m, v[i]); }
    red[tid] = m; __syncthreads();
    for (int s = 128; s > 0; s >>= 1) { if (tid < s) red[tid] = fmaxf(red[tid], red[tid + s]); __syncthreads(); }
    m = red[0]; __syncthreads();

    float sum = 0.f;
#pragma unroll
    for (int i = 0; i < 4; ++i) { v[i] = __expf(v[i] - m); sum += v[i]; }
    red[tid] = sum; __syncthreads();
    for (int s = 128; s > 0; s >>= 1) { if (tid < s) red[tid] += red[tid + s]; __syncthreads(); }
    float inv = 1.f / red[0];

#pragma unroll
    for (int i = 0; i < 4; ++i) {
        float wv = v[i] * inv;
        size_t idx = (size_t)row * SS + tid + i * 256;
        wts[idx] = wv;
        wbf[idx] = f2bf(wv);
    }
}

// ---------------------------------------------------------------------------
// GEMM3 (NN): context[b,t,e] = sum_s w[b,t,s]*enc[b,s,e]   (store bf16)
// enc is K-major, so stage its 32x128 tile in LDS transposed to [n][k].
// grid: (E/128, T/128, B)
// ---------------------------------------------------------------------------
__global__ void __launch_bounds__(256)
k_gemm_ctx(const unsigned short* __restrict__ w_bf,
           const unsigned short* __restrict__ enc_bf,
           unsigned short* __restrict__ ctx_bf) {
    __shared__ unsigned short ldsB[128 * 34];   // stride 34: 4B-aligned frag reads, spread banks
    const int b    = blockIdx.z;
    const int tid  = threadIdx.x;
    const int lane = tid & 31, w = tid >> 5;
    const int nBase0 = blockIdx.x * 128;
    const int mBase  = blockIdx.y * 128 + (w >> 1) * 32;
    const int nLoc   = (w & 1) * 64;
    const int hi = (lane >> 4) & 1, l15 = lane & 15;

    const unsigned short* A    = w_bf   + (size_t)b * TT * SS;
    const unsigned short* Bsrc = enc_bf + (size_t)b * SS * EE;

    // Hoisted row pointers for A fragments (immediate offsets in k loop).
    const unsigned short* ap[2];
#pragma unroll
    for (int mi = 0; mi < 2; ++mi)
        ap[mi] = A + (size_t)(mBase + mi * 16 + l15) * SS + hi * 16;
    // Hoisted LDS fragment bases.
    const unsigned int* lp[4];
#pragma unroll
    for (int ni = 0; ni < 4; ++ni)
        lp[ni] = (const unsigned int*)&ldsB[(nLoc + ni * 16 + l15) * 34 + hi * 16];
    // Staging: thread tid handles 8 uints (bf16 pairs); k = tid>>3, n2 = (tid&7)*16.
    // Consecutive tids read consecutive 8-uint chunks of one enc row -> coalesced.
    const int stg_k  = tid >> 3;         // 0..31 (256 threads / 8)
    const int stg_n2 = (tid & 7) * 16;   // 8 threads cover 128 cols, 16 cols each

    v8f zero = {0.f,0.f,0.f,0.f,0.f,0.f,0.f,0.f};
    v8f acc[2][4];
#pragma unroll
    for (int mi = 0; mi < 2; ++mi)
#pragma unroll
        for (int ni = 0; ni < 4; ++ni) acc[mi][ni] = zero;

    for (int kk = 0; kk < SS; kk += 32) {
        // stage B tile: enc rows kk..kk+31, cols nBase0..+127 -> LDS [n][k]
        {
            const unsigned int* g = (const unsigned int*)&Bsrc[(size_t)(kk + stg_k) * EE + nBase0 + stg_n2];
#pragma unroll
            for (int j = 0; j < 8; ++j) {
                unsigned int pair = g[j];
                int n = stg_n2 + 2 * j;
                ldsB[n * 34 + stg_k]       = (unsigned short)(pair & 0xFFFFu);
                ldsB[(n + 1) * 34 + stg_k] = (unsigned short)(pair >> 16);
            }
        }
        __syncthreads();

        FragBf a[2], bfr[4];
#pragma unroll
        for (int mi = 0; mi < 2; ++mi) {
            a[mi].q[0] = *(const uint4*)(ap[mi] + kk);
            a[mi].q[1] = *(const uint4*)(ap[mi] + kk + 8);
        }
#pragma unroll
        for (int ni = 0; ni < 4; ++ni)
#pragma unroll
            for (int i = 0; i < 8; ++i) bfr[ni].ui[i] = lp[ni][i];
#pragma unroll
        for (int ni = 0; ni < 4; ++ni)
#pragma unroll
            for (int mi = 0; mi < 2; ++mi)
                acc[mi][ni] = __builtin_amdgcn_wmma_f32_16x16x32_bf16(
                    false, a[mi].v, false, bfr[ni].v, (short)0, acc[mi][ni], false, false);
        __syncthreads();
    }

    unsigned short* out = ctx_bf + (size_t)b * TT * EE;
#pragma unroll
    for (int mi = 0; mi < 2; ++mi)
#pragma unroll
        for (int ni = 0; ni < 4; ++ni) {
            int n = nBase0 + nLoc + ni * 16 + l15;
#pragma unroll
            for (int r = 0; r < 8; ++r) {
                int m = mBase + mi * 16 + hi * 8 + r;
                out[(size_t)m * EE + n] = f2bf(acc[mi][ni][r]);
            }
        }
}

// ---------------------------------------------------------------------------
// GEMM4: h_tilde[b,t,d] = tanh( sum_c [ctx|hid][b,t,c] * W_out[d,c] )
// Concat handled by running the NT core twice (no materialized h_cat).
// grid: (D/128, T/128, B)
// ---------------------------------------------------------------------------
__global__ void __launch_bounds__(256)
k_gemm_out(const unsigned short* __restrict__ ctx_bf,
           const unsigned short* __restrict__ hid_bf,
           const unsigned short* __restrict__ Wo_bf,
           float* __restrict__ htilde) {
    const int b    = blockIdx.z;
    const int tid  = threadIdx.x;
    const int lane = tid & 31, w = tid >> 5;
    const int mBase = blockIdx.y * 128 + (w >> 1) * 32;
    const int nBase = blockIdx.x * 128 + (w & 1) * 64;
    const int hi = (lane >> 4) & 1, l15 = lane & 15;

    v8f zero = {0.f,0.f,0.f,0.f,0.f,0.f,0.f,0.f};
    v8f acc[2][4];
#pragma unroll
    for (int mi = 0; mi < 2; ++mi)
#pragma unroll
        for (int ni = 0; ni < 4; ++ni) acc[mi][ni] = zero;

    // context part: k in [0,E) of W_out rows
    gemm_nt_core(ctx_bf + (size_t)b * TT * EE, EE, Wo_bf,      CC, EE, mBase, nBase, lane, acc);
    // hidden part:  k in [E,E+D) of W_out rows
    gemm_nt_core(hid_bf + (size_t)b * TT * DD, DD, Wo_bf + EE, CC, DD, mBase, nBase, lane, acc);

    float* out = htilde + (size_t)b * TT * DD;
#pragma unroll
    for (int mi = 0; mi < 2; ++mi)
#pragma unroll
        for (int ni = 0; ni < 4; ++ni) {
            int n = nBase + ni * 16 + l15;
#pragma unroll
            for (int r = 0; r < 8; ++r) {
                int m = mBase + mi * 16 + hi * 8 + r;
                out[(size_t)m * DD + n] = tanhf(acc[mi][ni][r]);
            }
        }
}

// ---------------------------------------------------------------------------
// Launch. Workspace layout (bf16 staging, ~174 MB total):
//   hid_bf 32MB | enc_bf 32MB | Wa_bf 2MB | Wo_bf 4MB | proj_bf 32MB |
//   attnW_bf 32MB | ctx_bf 32MB
// d_out: h_tilde (B*T*D) | attn_weights (B*T*S) | attn_energies (B*T*S), fp32
// ---------------------------------------------------------------------------
extern "C" void kernel_launch(void* const* d_in, const int* in_sizes, int n_in,
                              void* d_out, int out_size, void* d_ws, size_t ws_size,
                              hipStream_t stream) {
    const float* hidden = (const float*)d_in[0];
    const float* enc    = (const float*)d_in[1];
    const float* W_attn = (const float*)d_in[2];
    const float* b_attn = (const float*)d_in[3];
    const float* W_out  = (const float*)d_in[4];

    float* out      = (float*)d_out;
    float* h_tilde  = out;
    float* attn_w   = out + (size_t)BB * TT * DD;
    float* energies = attn_w + (size_t)BB * TT * SS;

    unsigned char* ws = (unsigned char*)d_ws;
    size_t off = 0;
    unsigned short* hid_bf  = (unsigned short*)(ws + off); off += (size_t)BB * TT * DD * 2;
    unsigned short* enc_bf  = (unsigned short*)(ws + off); off += (size_t)BB * SS * EE * 2;
    unsigned short* Wa_bf   = (unsigned short*)(ws + off); off += (size_t)DD * EE * 2;
    unsigned short* Wo_bf   = (unsigned short*)(ws + off); off += (size_t)DD * CC * 2;
    unsigned short* proj_bf = (unsigned short*)(ws + off); off += (size_t)BB * SS * DD * 2;
    unsigned short* w_bf    = (unsigned short*)(ws + off); off += (size_t)BB * TT * SS * 2;
    unsigned short* ctx_bf  = (unsigned short*)(ws + off); off += (size_t)BB * TT * EE * 2;
    (void)ws_size; (void)in_sizes; (void)n_in; (void)out_size;

    // 1) fp32 -> bf16 staging
    k_cvt_bf16<<<2048, 256, 0, stream>>>(hidden, hid_bf, (size_t)BB * TT * DD);
    k_cvt_bf16<<<2048, 256, 0, stream>>>(enc,    enc_bf, (size_t)BB * SS * EE);
    k_cvt_bf16<<<512,  256, 0, stream>>>(W_attn, Wa_bf,  (size_t)DD * EE);
    k_cvt_bf16<<<512,  256, 0, stream>>>(W_out,  Wo_bf,  (size_t)DD * CC);

    dim3 blk(256);
    // 2) proj = enc @ W_attn^T + b
    k_gemm_proj<<<dim3(DD / 128, SS / 128, BB), blk, 0, stream>>>(enc_bf, Wa_bf, b_attn, proj_bf);
    // 3) energies = hidden @ proj^T  (output #3)
    k_gemm_energy<<<dim3(SS / 128, TT / 128, BB), blk, 0, stream>>>(hid_bf, proj_bf, energies);
    // 4) softmax -> attn_weights (output #2) + bf16 copy
    k_softmax<<<BB * TT, blk, 0, stream>>>(energies, attn_w, w_bf);
    // 5) context = weights @ enc
    k_gemm_ctx<<<dim3(EE / 128, TT / 128, BB), blk, 0, stream>>>(w_bf, enc_bf, ctx_bf);
    // 6) h_tilde = tanh([ctx|hid] @ W_out^T)  (output #1)
    k_gemm_out<<<dim3(DD / 128, TT / 128, BB), blk, 0, stream>>>(ctx_bf, hid_bf, Wo_bf, h_tilde);
}
